// NTXentLoss_89575837925894
// MI455X (gfx1250) — matrix operand order repaired and checked by reading, not compiled, not run on hardware
//
#include <hip/hip_runtime.h>
#include <hip/hip_bf16.h>

#define DIM 256
#define N2  8192
#define NUMROWS (13 * 8192) /* 106496 flat 256-float rows in emb */

typedef __attribute__((ext_vector_type(16))) __bf16 v16bf;
typedef __attribute__((ext_vector_type(8)))  float  v8f;

union FragU { v16bf v; uint4 u[2]; };

__device__ __forceinline__ float wave_allsum(float v) {
    v += __shfl_xor(v, 1);
    v += __shfl_xor(v, 2);
    v += __shfl_xor(v, 4);
    v += __shfl_xor(v, 8);
    v += __shfl_xor(v, 16);
    return v;
}

// ---------------------------------------------------------------------------
// Kernel 1: row norms for all 13*8192 rows; bf16 normalized z0 (first 8192
// rows) for the WMMA GEMM; zero the per-row exp-sum accumulators.
// Pure bandwidth: 109 MB in -> ~5 us at 23.3 TB/s.
// ---------------------------------------------------------------------------
__global__ void __launch_bounds__(256) norm_kernel(
    const float* __restrict__ emb, unsigned short* __restrict__ Zb,
    float* __restrict__ invn, float* __restrict__ rowsum) {
    const int lane = threadIdx.x & 31;
    const int row  = blockIdx.x * 8 + (threadIdx.x >> 5); // 1 wave = 1 row
    const float4* src = (const float4*)(emb + (size_t)row * DIM) + lane * 2;
    float4 x0 = src[0], x1 = src[1];
    float s = x0.x*x0.x + x0.y*x0.y + x0.z*x0.z + x0.w*x0.w
            + x1.x*x1.x + x1.y*x1.y + x1.z*x1.z + x1.w*x1.w;
    s = wave_allsum(s);
    const float inv = 1.0f / fmaxf(sqrtf(s), 1e-8f);
    if (lane == 0) invn[row] = inv;
    if (row < N2) {
        union { __bf16 h[8]; uint4 u; } pk;
        pk.h[0] = (__bf16)(x0.x * inv); pk.h[1] = (__bf16)(x0.y * inv);
        pk.h[2] = (__bf16)(x0.z * inv); pk.h[3] = (__bf16)(x0.w * inv);
        pk.h[4] = (__bf16)(x1.x * inv); pk.h[5] = (__bf16)(x1.y * inv);
        pk.h[6] = (__bf16)(x1.z * inv); pk.h[7] = (__bf16)(x1.w * inv);
        *((uint4*)(Zb + (size_t)row * DIM) + lane) = pk.u;
        if (lane == 0) rowsum[row] = 0.0f;
    }
}

// ---------------------------------------------------------------------------
// Kernel 2: fused Gram-matrix + softmax denominator.
// Grid (64,64): each 256-thread WG computes a 128x128 tile of S = Zb*Zb^T
// with v_wmma_f32_16x16x32_bf16 (full K=256 in 128 KB dynamic LDS), then in
// the epilogue accumulates exp(10*S[n,j]) into rowsum[n] for the allowed
// columns and captures the positive logit S[n, n^4096].
// ---------------------------------------------------------------------------
__global__ void __launch_bounds__(256) simi_kernel(
    const unsigned short* __restrict__ Zb, const int* __restrict__ rptr,
    float* __restrict__ rowsum, float* __restrict__ posval) {
    extern __shared__ char smem[];
    unsigned short* As = (unsigned short*)smem;     // 128 x 256 bf16
    unsigned short* Bs = As + 128 * DIM;            // 128 x 256 bf16

    const int tid   = threadIdx.x;
    const int lane  = tid & 31;
    const int wav   = tid >> 5;
    const int half  = lane >> 4;
    const int lr    = lane & 15;
    const int rbase = blockIdx.x * 128;
    const int cbase = blockIdx.y * 128;
    const int r     = rptr[0];                      // random_num (=512)

    // Both tiles are contiguous 64 KB slabs of Zb (row-major, K-contiguous).
    const uint4* ga = (const uint4*)(Zb + (size_t)rbase * DIM);
    const uint4* gb = (const uint4*)(Zb + (size_t)cbase * DIM);
    __builtin_prefetch((const char*)gb + tid * 64, 0, 1); // global_prefetch_b8
    uint4* la = (uint4*)As;
    uint4* lb = (uint4*)Bs;
    for (int i = tid; i < 4096; i += 256) la[i] = ga[i];
    for (int i = tid; i < 4096; i += 256) lb[i] = gb[i];
    __syncthreads();

    v8f acc[8];
    const v8f vzero = {0.f, 0.f, 0.f, 0.f, 0.f, 0.f, 0.f, 0.f};
#pragma unroll
    for (int t = 0; t < 8; ++t) acc[t] = vzero;

    const int nrow0 = rbase + 16 * wav; // this wave owns rows [nrow0, nrow0+16)

#pragma unroll
    for (int k0 = 0; k0 < DIM; k0 += 32) {
        // A 16x32 bf16 frag: lanes 0-15 -> K {k0..k0+7, k0+16..k0+23},
        // lanes 16-31 -> K {k0+8..k0+15, k0+24..k0+31} (ISA 7.12.2).
        FragU a;
        const unsigned short* ap = As + (16 * wav + lr) * DIM + k0 + half * 8;
        a.u[0] = *(const uint4*)ap;
        a.u[1] = *(const uint4*)(ap + 16);
#pragma unroll
        for (int t = 0; t < 8; ++t) {
            // B 32x16 bf16 frag: lane = column, same K split as A.
            FragU b;
            const unsigned short* bp = Bs + (16 * t + lr) * DIM + k0 + half * 8;
            b.u[0] = *(const uint4*)bp;
            b.u[1] = *(const uint4*)(bp + 16);
            acc[t] = __builtin_amdgcn_wmma_f32_16x16x32_bf16(
                false, a.v, false, b.v, (short)0, acc[t], false, false);
        }
    }

    // Epilogue: C/D layout -> element (t,i): n = nrow0 + 8*half + i,
    // j = cbase + 16*t + (lane&15).
    float rowAcc[8];
#pragma unroll
    for (int i = 0; i < 8; ++i) rowAcc[i] = 0.0f;

#pragma unroll
    for (int t = 0; t < 8; ++t) {
        const int colbase = cbase + 16 * t;
        const int j       = colbase + lr;
        const bool tDiag  = (colbase == nrow0);            // self column possible
        const bool tPos   = (colbase == (nrow0 ^ 4096));   // partner column possible
        const bool tDrop  = (colbase + 16 > r) && (colbase < r + 26); // dropped window
#pragma unroll
        for (int i = 0; i < 8; ++i) {
            const int   n   = nrow0 + 8 * half + i;
            const float val = acc[t][i];
            bool excl = tDiag && (j == n);                 // exclude self-similarity
            if (tDrop) {
                const int a = n & 4095;
                const int b = a + 4096;
                if (j != a && j != b) {                    // self/partner not in MASK2
                    const int kidx = j - (j > a) - (j > b);
                    excl |= (kidx >= r) && (kidx < r + 24);
                }
            }
            if (!excl) rowAcc[i] += __expf(val * 10.0f);   // /TEMP
            if (tPos && (j == (n ^ 4096))) posval[n] = val;
        }
    }

    // Reduce across the 16 lanes sharing each row, then one atomic per row.
#pragma unroll
    for (int i = 0; i < 8; ++i) {
        float v = rowAcc[i];
        v += __shfl_xor(v, 1);
        v += __shfl_xor(v, 2);
        v += __shfl_xor(v, 4);
        v += __shfl_xor(v, 8);
        if (lr == 0) atomicAdd(&rowsum[nrow0 + 8 * half + i], v);
    }
}

// ---------------------------------------------------------------------------
// Kernel 3: 24 hard-negative logits per row in f32 (matches reference math).
// zp[k][j] is emb flat row 8192 + 8192*k + j; row n uses j = a and a+4096.
// 100 MFLOP total -> VALU noise.
// ---------------------------------------------------------------------------
__global__ void __launch_bounds__(256) hard_kernel(
    const float* __restrict__ emb, const float* __restrict__ invn,
    float* __restrict__ rowsum) {
    const int lane = threadIdx.x & 31;
    const int n    = blockIdx.x * 8 + (threadIdx.x >> 5);
    const int a    = n & 4095;
    const float4* z0p = (const float4*)(emb + (size_t)n * DIM) + lane * 2;
    float4 z0a = z0p[0], z0b = z0p[1];
    const float inv0 = invn[n];
    z0a.x *= inv0; z0a.y *= inv0; z0a.z *= inv0; z0a.w *= inv0;
    z0b.x *= inv0; z0b.y *= inv0; z0b.z *= inv0; z0b.w *= inv0;
    float hsum = 0.0f;
#pragma unroll
    for (int k = 0; k < 12; ++k) {
#pragma unroll
        for (int p = 0; p < 2; ++p) {
            const int rrow = 8192 * (k + 1) + a + 4096 * p;
            const float4* zp = (const float4*)(emb + (size_t)rrow * DIM) + lane * 2;
            const float4 y0 = zp[0], y1 = zp[1];
            float d = z0a.x*y0.x + z0a.y*y0.y + z0a.z*y0.z + z0a.w*y0.w
                    + z0b.x*y1.x + z0b.y*y1.y + z0b.z*y1.z + z0b.w*y1.w;
            d = wave_allsum(d);
            hsum += __expf(d * invn[rrow] * 10.0f);
        }
    }
    if (lane == 0) atomicAdd(&rowsum[n], hsum);
}

// ---------------------------------------------------------------------------
// Kernel 4: loss = mean_n( log(sum_exp_n) - pos_n/T ).
// ---------------------------------------------------------------------------
__global__ void __launch_bounds__(256) loss_kernel(
    const float* __restrict__ rowsum, const float* __restrict__ posval,
    float* __restrict__ out) {
    __shared__ float sred[8];
    const int tid = threadIdx.x;
    float s = 0.0f;
    for (int n = tid; n < N2; n += 256)
        s += __logf(rowsum[n]) - posval[n] * 10.0f;
    s = wave_allsum(s);
    if ((tid & 31) == 0) sred[tid >> 5] = s;
    __syncthreads();
    if (tid == 0) {
        float t = 0.0f;
        for (int w = 0; w < 8; ++w) t += sred[w];
        out[0] = t / (float)N2;
    }
}

extern "C" void kernel_launch(void* const* d_in, const int* in_sizes, int n_in,
                              void* d_out, int out_size, void* d_ws, size_t ws_size,
                              hipStream_t stream) {
    const float* emb  = (const float*)d_in[0]; // (26,4096,256) f32
    const int*   rnum = (const int*)d_in[2];   // random_num (512)

    char* ws = (char*)d_ws;
    unsigned short* Zb = (unsigned short*)ws;                       // 4 MiB bf16 z0
    float* invn   = (float*)(ws + (size_t)N2 * DIM * 2);            // 13*8192 f32
    float* rowsum = invn + NUMROWS;                                 // 8192 f32
    float* posval = rowsum + N2;                                    // 8192 f32
    float* out    = (float*)d_out;

    norm_kernel<<<NUMROWS / 8, 256, 0, stream>>>(emb, Zb, invn, rowsum);
    simi_kernel<<<dim3(64, 64), 256, 2 * 128 * DIM * sizeof(unsigned short), stream>>>(
        Zb, rnum, rowsum, posval);
    hard_kernel<<<N2 / 8, 256, 0, stream>>>(emb, invn, rowsum);
    loss_kernel<<<1, 256, 0, stream>>>(rowsum, posval, out);
}
